// QueueMemoryBank_13984413515821
// MI455X (gfx1250) — matrix operand order repaired and checked by reading, not compiled, not run on hardware
//
#include <hip/hip_runtime.h>
#include <hip/hip_bf16.h>
#include <cstdint>
#include <cstddef>

typedef __attribute__((ext_vector_type(16))) __bf16       v16bf;
typedef __attribute__((ext_vector_type(8)))  float        v8f;
typedef __attribute__((ext_vector_type(8)))  unsigned int v8u;

#define BQ     256      // batch (query) rows
#define DDIM   1024     // feature dim
#define MMEM   65536    // memory slots
#define KSTEP  32       // bf16 WMMA K per step
#define NSLAB  256      // N columns per workgroup
#define NT     16       // 16x16 C tiles per wave (N direction)
#define WGSIZE 512      // 16 waves
#define LDSROWB 80      // LDS bytes per 32-elem bf16 row (64B + 16B pad, keeps 16B align)
#define CHUNKS 32       // split-K chunks over MMEM in kernel 3
#define MCHUNK (MMEM / CHUNKS)
#define DSLICE 256      // D columns per workgroup in kernel 3

// ---- helpers -------------------------------------------------------------

// round-to-nearest-even f32 -> bf16, packed pair (a in [15:0], b in [31:16])
__device__ __forceinline__ unsigned int pk_bf16x2(float a, float b) {
  unsigned ua = __builtin_bit_cast(unsigned, a);
  unsigned ub = __builtin_bit_cast(unsigned, b);
  ua += 0x7FFFu + ((ua >> 16) & 1u);
  ub += 0x7FFFu + ((ub >> 16) & 1u);
  return (ua >> 16) | (ub & 0xFFFF0000u);
}

// Build a 16x32 bf16 A-fragment for this lane directly from a global f32 row.
// Layout (16-bit A, matches B): half-lane h: V0-3 hold K=16h..16h+7, V4-7 hold K=16h+8..16h+15,
// even K in [15:0] of each VGPR.
__device__ __forceinline__ v16bf load_afrag_f32(const float* __restrict__ rowk, int h) {
  const float4* p0 = reinterpret_cast<const float4*>(rowk + 16 * h);
  const float4* p1 = reinterpret_cast<const float4*>(rowk + 16 * h + 8);
  float4 f0 = p0[0], f1 = p0[1], f2 = p1[0], f3 = p1[1];
  v8u u;
  u[0] = pk_bf16x2(f0.x, f0.y); u[1] = pk_bf16x2(f0.z, f0.w);
  u[2] = pk_bf16x2(f1.x, f1.y); u[3] = pk_bf16x2(f1.z, f1.w);
  u[4] = pk_bf16x2(f2.x, f2.y); u[5] = pk_bf16x2(f2.z, f2.w);
  u[6] = pk_bf16x2(f3.x, f3.y); u[7] = pk_bf16x2(f3.z, f3.w);
  return __builtin_bit_cast(v16bf, u);
}

// Load a 32x16 bf16 B-fragment from LDS. Tile stored as [n][k] bf16, row stride
// LDSROWB bytes. Layout used here (16-bit B, same k order as A): half-lane h holds K=16h..16h+15 across V0-7,
// i.e. 32 contiguous bytes per lane -> two 16B LDS loads.
__device__ __forceinline__ v16bf load_bfrag_lds(const char* lds, int row, int h) {
  const uint4* p = reinterpret_cast<const uint4*>(lds + row * LDSROWB + 32 * h);
  uint4 lo = p[0], hi = p[1];
  v8u u;
  u[0] = lo.x; u[1] = lo.y; u[2] = lo.z; u[3] = lo.w;
  u[4] = hi.x; u[5] = hi.y; u[6] = hi.z; u[7] = hi.w;
  return __builtin_bit_cast(v16bf, u);
}

// ---- kernel 1: logits = x @ mem^T (bf16 WMMA, f32 accum) ------------------
// Grid: MMEM/NSLAB blocks. Each block: all 256 query rows x 256 mem rows, so
// mem is streamed from HBM exactly once across the grid. Global->register
// prefetch double-buffers the staging so HBM loads overlap the WMMAs.
__global__ __launch_bounds__(WGSIZE) void k_logits(const float* __restrict__ x,
                                                   const float* __restrict__ mem,
                                                   float* __restrict__ logits) {
  __shared__ __align__(16) char blds[NSLAB * LDSROWB];  // 20 KB
  const int tid  = threadIdx.x;
  const int lane = tid & 31;
  const int wave = tid >> 5;
  const int n    = lane & 15;
  const int h    = lane >> 4;
  const int n0   = blockIdx.x * NSLAB;
  const float* xrow = x + (size_t)(wave * 16 + n) * DDIM;  // A row for this lane

  // loop-invariant staging addresses (per-thread slice of the 256x32 tile)
  const float* gsrc[4];
  char*        ldst[4];
#pragma unroll
  for (int p = 0; p < 4; ++p) {
    int flat = p * WGSIZE + tid;   // float4 index within 256x32 tile
    int r = flat >> 3;
    int q = flat & 7;
    gsrc[p] = mem + (size_t)(n0 + r) * DDIM + q * 4;
    ldst[p] = blds + r * LDSROWB + q * 8;
  }

  v8f acc[NT] = {};

  float4 pf[4];
#pragma unroll
  for (int p = 0; p < 4; ++p) pf[p] = *reinterpret_cast<const float4*>(gsrc[p]);

  for (int k0 = 0; k0 < DDIM; k0 += KSTEP) {
    // commit prefetched tile to LDS as bf16
#pragma unroll
    for (int p = 0; p < 4; ++p) {
      uint2 u;
      u.x = pk_bf16x2(pf[p].x, pf[p].y);
      u.y = pk_bf16x2(pf[p].z, pf[p].w);
      *reinterpret_cast<uint2*>(ldst[p]) = u;
    }
    __syncthreads();

    // issue next tile's global loads; they complete under the WMMAs below
    if (k0 + KSTEP < DDIM) {
#pragma unroll
      for (int p = 0; p < 4; ++p)
        pf[p] = *reinterpret_cast<const float4*>(gsrc[p] + k0 + KSTEP);
    }

    v16bf afrag = load_afrag_f32(xrow + k0, h);
#pragma unroll
    for (int t = 0; t < NT; ++t) {
      v16bf bfrag = load_bfrag_lds(blds, t * 16 + n, h);
      acc[t] = __builtin_amdgcn_wmma_f32_16x16x32_bf16(
          false, afrag, false, bfrag, (short)0, acc[t], false, false);
    }
    __syncthreads();
  }

  // C layout: VGPR v -> M = v + 8h, N = lane%16
  const int mb = wave * 16;
#pragma unroll
  for (int t = 0; t < NT; ++t) {
#pragma unroll
    for (int v = 0; v < 8; ++v) {
      logits[(size_t)(mb + h * 8 + v) * MMEM + n0 + t * 16 + n] = acc[t][v];
    }
  }
}

// ---- kernel 2: in-place row softmax over 65536 (f32, L2-resident row) -----
__global__ __launch_bounds__(512) void k_softmax(float* __restrict__ logits) {
  __shared__ float red[512];
  const int tid = threadIdx.x;
  float* row = logits + (size_t)blockIdx.x * MMEM;

  float m = -3.402823466e38f;
  for (int i = tid; i < MMEM; i += 512) m = fmaxf(m, row[i]);
  red[tid] = m;
  __syncthreads();
  for (int s = 256; s > 0; s >>= 1) {
    if (tid < s) red[tid] = fmaxf(red[tid], red[tid + s]);
    __syncthreads();
  }
  m = red[0];
  __syncthreads();

  float sum = 0.f;
  for (int i = tid; i < MMEM; i += 512) sum += __expf(row[i] - m);
  red[tid] = sum;
  __syncthreads();
  for (int s = 256; s > 0; s >>= 1) {
    if (tid < s) red[tid] += red[tid + s];
    __syncthreads();
  }
  const float inv = 1.f / red[0];
  __syncthreads();

  for (int i = tid; i < MMEM; i += 512) row[i] = __expf(row[i] - m) * inv;
}

// ---- kernel 3: partial[c] = attn[:, chunk_c] @ mem[chunk_c, dslice] -------
// Grid: CHUNKS*4 blocks; block (c, s) covers all 256 rows x 256 d-columns.
// Same register-prefetch pipelining as kernel 1.
__global__ __launch_bounds__(WGSIZE) void k_out_partial(const float* __restrict__ attn,
                                                        const float* __restrict__ mem,
                                                        float* __restrict__ partial) {
  __shared__ __align__(16) char blds[DSLICE * LDSROWB];  // 20 KB
  const int tid  = threadIdx.x;
  const int lane = tid & 31;
  const int wave = tid >> 5;
  const int n    = lane & 15;
  const int h    = lane >> 4;
  const int c    = blockIdx.x >> 2;
  const int d0   = (blockIdx.x & 3) * DSLICE;
  const int m0b  = c * MCHUNK;
  const float* attnrow = attn + (size_t)(wave * 16 + n) * MMEM + m0b;

  // loop-invariant staging addresses: thread p-slot handles (col, k-pair k2)
  const float* gsrc[8];
  char*        ldst[8];
#pragma unroll
  for (int p = 0; p < 8; ++p) {
    int flat = p * WGSIZE + tid;     // 4096 (col,kpair) slots
    int k2  = flat >> 8;             // 0..15
    int col = flat & 255;
    gsrc[p] = mem + (size_t)(m0b + 2 * k2) * DDIM + d0 + col;
    ldst[p] = blds + col * LDSROWB + k2 * 4;
  }

  v8f acc[NT] = {};

  float pfa[8], pfb[8];
#pragma unroll
  for (int p = 0; p < 8; ++p) { pfa[p] = gsrc[p][0]; pfb[p] = gsrc[p][DDIM]; }

  for (int kk = 0; kk < MCHUNK; kk += KSTEP) {
    // commit prefetched tile to LDS, transposed: lds[col][k] (k-pairs packed)
#pragma unroll
    for (int p = 0; p < 8; ++p)
      *reinterpret_cast<unsigned int*>(ldst[p]) = pk_bf16x2(pfa[p], pfb[p]);
    __syncthreads();

    // prefetch next 32-row slab of mem while WMMAs run
    if (kk + KSTEP < MCHUNK) {
      const size_t off = (size_t)(kk + KSTEP) * DDIM;
#pragma unroll
      for (int p = 0; p < 8; ++p) {
        pfa[p] = gsrc[p][off];
        pfb[p] = gsrc[p][off + DDIM];
      }
    }

    v16bf afrag = load_afrag_f32(attnrow + kk, h);
#pragma unroll
    for (int t = 0; t < NT; ++t) {
      v16bf bfrag = load_bfrag_lds(blds, t * 16 + n, h);
      acc[t] = __builtin_amdgcn_wmma_f32_16x16x32_bf16(
          false, afrag, false, bfrag, (short)0, acc[t], false, false);
    }
    __syncthreads();
  }

  float* pout = partial + (size_t)c * (BQ * DDIM);
  const int mb = wave * 16;
#pragma unroll
  for (int t = 0; t < NT; ++t) {
#pragma unroll
    for (int v = 0; v < 8; ++v) {
      pout[(size_t)(mb + h * 8 + v) * DDIM + d0 + t * 16 + n] = acc[t][v];
    }
  }
}

// ---- kernel 4: deterministic split-K reduction ----------------------------
__global__ __launch_bounds__(256) void k_reduce(const float* __restrict__ partial,
                                                float* __restrict__ out) {
  const int e = blockIdx.x * 256 + threadIdx.x;  // 0..BQ*DDIM-1
  float a = 0.f;
#pragma unroll
  for (int c = 0; c < CHUNKS; ++c) a += partial[(size_t)c * (BQ * DDIM) + e];
  out[e] = a;
}

// ---- launcher -------------------------------------------------------------
extern "C" void kernel_launch(void* const* d_in, const int* in_sizes, int n_in,
                              void* d_out, int out_size, void* d_ws, size_t ws_size,
                              hipStream_t stream) {
  const float* x   = (const float*)d_in[0];   // (256, 1, 1024) f32
  const float* mem = (const float*)d_in[1];   // (65536, 1024) f32

  // workspace layout: [0, 64MB) logits (f32, softmaxed in place -> attn)
  //                   [64MB, 96MB) split-K partials (f32)
  float* logits  = (float*)d_ws;
  float* partial = (float*)((char*)d_ws + (size_t)BQ * MMEM * sizeof(float));

  k_logits<<<MMEM / NSLAB, WGSIZE, 0, stream>>>(x, mem, logits);
  k_softmax<<<BQ, 512, 0, stream>>>(logits);
  k_out_partial<<<CHUNKS * 4, WGSIZE, 0, stream>>>(logits, mem, partial);
  k_reduce<<<(BQ * DDIM) / 256, 256, 0, stream>>>(partial, (float*)d_out);
}
